// TransConvLayer_27685359190150
// MI455X (gfx1250) — matrix operand order repaired and checked by reading, not compiled, not run on hardware
//
#include <hip/hip_runtime.h>
#include <hip/hip_bf16.h>

typedef _Float16 f16;
typedef __attribute__((ext_vector_type(16))) _Float16 v16h;
typedef __attribute__((ext_vector_type(8)))  _Float16 v8h;
typedef __attribute__((ext_vector_type(8)))  float    v8f;
typedef __attribute__((ext_vector_type(4)))  float    f32x4;
typedef __attribute__((ext_vector_type(4)))  int      v4i;

#define KCURV 1.0f
#define EPSV  1e-6f

constexpr int B_ = 8, N_ = 16384, C_ = 64, H_ = 4, D_ = 63;

constexpr int WP = 72;   // f16 pitch for Bt-format weight tiles (K contiguous)
constexpr int TP = 40;   // f16 pitch for token-transposed tiles (32 tokens + pad)
constexpr int RP = 72;   // f32 pitch for raw staging buffers

constexpr int KTV_ELEMS  = B_ * H_ * 64 * 64;
constexpr int KSUM_ELEMS = B_ * H_ * 64;

// ------------------------- async global->LDS staging -------------------------
#if defined(__HIP_DEVICE_COMPILE__) && \
    __has_builtin(__builtin_amdgcn_global_load_async_to_lds_b128) && \
    __has_builtin(__builtin_amdgcn_s_wait_asynccnt)
#define USE_ASYNC_LDS 1
#else
#define USE_ASYNC_LDS 0
#endif

#if USE_ASYNC_LDS
typedef __attribute__((address_space(1))) v4i as1_v4i;
typedef __attribute__((address_space(3))) v4i as3_v4i;
#endif

__device__ __forceinline__ void stage_g2l(const float* __restrict__ g,
                                          float* __restrict__ l,
                                          int nfloats, int tid, int nthr) {
#if USE_ASYNC_LDS
  for (int i = tid; i < (nfloats >> 2); i += nthr) {
    as1_v4i* gp = (as1_v4i*)(unsigned long long)(const void*)(g + i * 4);
    as3_v4i* lp = (as3_v4i*)(unsigned int)(unsigned long long)(void*)(l + i * 4);
    __builtin_amdgcn_global_load_async_to_lds_b128(gp, lp, 0, 0);
  }
  __builtin_amdgcn_s_wait_asynccnt(0);
#else
  for (int i = tid; i < nfloats; i += nthr) l[i] = g[i];
#endif
}

// ------------------------------- WMMA helpers --------------------------------
__device__ __forceinline__ v8f vzero() {
  v8f a;
#pragma unroll
  for (int j = 0; j < 8; ++j) a[j] = 0.0f;
  return a;
}

__device__ __forceinline__ v8f wmma16(v16h a, v16h b, v8f c) {
  return __builtin_amdgcn_wmma_f32_16x16x32_f16(false, a, false, b, (short)0, c,
                                                false, false);
}

__device__ __forceinline__ v16h cat8(v8h lo, v8h hi) {
  v16h r;
#pragma unroll
  for (int i = 0; i < 8; ++i) { r[i] = lo[i]; r[i + 8] = hi[i]; }
  return r;
}

// A-tile (16xK slice, K=32) from row-major [row][k] f16 LDS matrix.
__device__ __forceinline__ v16h load_lA(const f16* m, int pitch, int r0, int k0,
                                        int lane) {
  const f16* p = m + (r0 + (lane & 15)) * pitch + k0 + ((lane >> 4) * 8);
  return cat8(*(const v8h*)p, *(const v8h*)(p + 16));
}

// B-tile (Kx16 slice, K=32) from Bt-format [n][k] f16 LDS matrix.
__device__ __forceinline__ v16h load_lB(const f16* bt, int pitch, int n0, int k0,
                                        int lane) {
  const f16* p = bt + (n0 + (lane & 15)) * pitch + k0 + ((lane >> 4) * 8);
  return cat8(*(const v8h*)p, *(const v8h*)(p + 16));
}

// A-tile built directly from global fp32 row-major activations [token][64].
__device__ __forceinline__ v16h load_gA(const float* rowp, int koff) {
  f32x4 a0 = *(const f32x4*)(rowp + koff);
  f32x4 a1 = *(const f32x4*)(rowp + koff + 4);
  f32x4 b0 = *(const f32x4*)(rowp + koff + 16);
  f32x4 b1 = *(const f32x4*)(rowp + koff + 20);
  v16h r;
#pragma unroll
  for (int i = 0; i < 4; ++i) {
    r[i]      = (f16)a0[i];
    r[4 + i]  = (f16)a1[i];
    r[8 + i]  = (f16)b0[i];
    r[12 + i] = (f16)b1[i];
  }
  return r;
}

// ------------------------------- zero kernel ---------------------------------
__global__ void zero_ws_kernel(float* p, int n) {
  int i = blockIdx.x * blockDim.x + threadIdx.x;
  if (i < n) p[i] = 0.0f;
}

// =============================== Kernel A ====================================
// Per (b,h,chunk): k/v projections (WMMA), phi_k, ktv += phi_k^T x v (WMMA),
// ksum accumulation. Reduced into workspace with float atomics.
constexpr int KA_THREADS = 128;
constexpr int KA_WAVES   = 4;
constexpr int CHUNK      = 512;              // tokens per block
constexpr int WTOK       = CHUNK / KA_WAVES; // 128 tokens per wave
constexpr int AITERS     = WTOK / 32;        // 4

constexpr size_t A_WKT   = 0;                          // f16[64*WP]
constexpr size_t A_WVT   = A_WKT + 64 * WP * 2;        // f16[64*WP]
constexpr size_t A_BK    = A_WVT + 64 * WP * 2;        // float[64]
constexpr size_t A_BV    = A_BK + 256;                 // float[64]
constexpr size_t A_KSUM  = A_BV + 256;                 // float[64]
constexpr size_t A_STAGE = A_KSUM + 256;               // float[64*64]
constexpr size_t A_WAVE0 = A_STAGE + 64 * 64 * 4;
constexpr size_t AW_KS   = 0;                          // float[32*RP]
constexpr size_t AW_PHIT = AW_KS + 32 * RP * 4;        // f16[64*TP]
constexpr size_t AW_VT   = AW_PHIT + 64 * TP * 2;      // f16[64*TP]
constexpr size_t AW_SZ   = AW_VT + 64 * TP * 2;
constexpr size_t SMEM_A  = A_WAVE0 + KA_WAVES * AW_SZ;

__global__ __launch_bounds__(KA_THREADS)
void kv_ktv_kernel(const float* __restrict__ src,
                   const float* __restrict__ Wk, const float* __restrict__ bk_g,
                   const float* __restrict__ Wv, const float* __restrict__ bv_g,
                   const float* __restrict__ nscale,
                   float* __restrict__ g_ktv, float* __restrict__ g_ksum) {
  extern __shared__ char smem[];
  f16*   WkT    = (f16*)(smem + A_WKT);
  f16*   WvT    = (f16*)(smem + A_WVT);
  float* bk     = (float*)(smem + A_BK);
  float* bv     = (float*)(smem + A_BV);
  float* ksum   = (float*)(smem + A_KSUM);
  float* wstage = (float*)(smem + A_STAGE);

  const int tid  = threadIdx.x;
  const int lane = tid & 31;
  const int w    = tid >> 5;
  char* wb       = smem + A_WAVE0 + (size_t)w * AW_SZ;
  float* ksraw   = (float*)(wb + AW_KS);
  f16*   phiT    = (f16*)(wb + AW_PHIT);
  f16*   vT      = (f16*)(wb + AW_VT);

  const int nchunks = N_ / CHUNK;
  const int chunk = blockIdx.x % nchunks;
  const int bh    = blockIdx.x / nchunks;
  const int h     = bh % H_;
  const int b     = bh / H_;
  const float inv = 1.0f / fabsf(nscale[0]);

  // ---- stage + convert Wk (async LDS path if available) ----
  stage_g2l(Wk + (size_t)h * D_ * C_, wstage, D_ * C_, tid, KA_THREADS);
  __syncthreads();
  for (int i = tid; i < 64 * 64; i += KA_THREADS) {
    int o = i >> 6, c = i & 63;
    WkT[o * WP + c] = (f16)((o < D_) ? wstage[o * C_ + c] : 0.0f);
  }
  __syncthreads();
  stage_g2l(Wv + (size_t)h * D_ * C_, wstage, D_ * C_, tid, KA_THREADS);
  __syncthreads();
  for (int i = tid; i < 64 * 64; i += KA_THREADS) {
    int o = i >> 6, c = i & 63;
    WvT[o * WP + c] = (f16)((o < D_) ? wstage[o * C_ + c] : 0.0f);
  }
  if (tid < 64) {
    bk[tid]   = (tid < D_) ? bk_g[h * D_ + tid] : 0.0f;
    bv[tid]   = (tid < D_) ? bv_g[h * D_ + tid] : 0.0f;
    ksum[tid] = 0.0f;
  }
  __syncthreads();

  v8f acc[16];
#pragma unroll
  for (int i = 0; i < 16; ++i) acc[i] = vzero();

  const int tok0w = chunk * CHUNK + w * WTOK;
  for (int it = 0; it < AITERS; ++it) {
    const int tok0 = tok0w + it * 32;
    const float* xbase = src + ((size_t)b * N_ + tok0) * C_;
    __builtin_prefetch(xbase + 32 * C_ + lane * 8, 0, 1);

    // ---- stage 1: k/v projections via WMMA ----
    v16h A[2][2];
#pragma unroll
    for (int rt = 0; rt < 2; ++rt) {
      const float* rp = xbase + (size_t)(rt * 16 + (lane & 15)) * C_;
#pragma unroll
      for (int ks = 0; ks < 2; ++ks)
        A[rt][ks] = load_gA(rp, ks * 32 + ((lane >> 4) * 8));
    }
#pragma unroll
    for (int ct = 0; ct < 4; ++ct) {
      v16h Bk0 = load_lB(WkT, WP, ct * 16, 0, lane);
      v16h Bk1 = load_lB(WkT, WP, ct * 16, 32, lane);
      v16h Bv0 = load_lB(WvT, WP, ct * 16, 0, lane);
      v16h Bv1 = load_lB(WvT, WP, ct * 16, 32, lane);
      const int n = ct * 16 + (lane & 15);
      const float bkn = bk[n], bvn = bv[n];
#pragma unroll
      for (int rt = 0; rt < 2; ++rt) {
        v8f dk = vzero();
        dk = wmma16(A[rt][0], Bk0, dk);
        dk = wmma16(A[rt][1], Bk1, dk);
        v8f dv = vzero();
        dv = wmma16(A[rt][0], Bv0, dv);
        dv = wmma16(A[rt][1], Bv1, dv);
        const int mb = rt * 16 + ((lane >> 4) * 8);
#pragma unroll
        for (int j = 0; j < 8; ++j) {
          ksraw[(mb + j) * RP + n] = dk[j] + bkn;        // raw ks, row-major
          vT[n * TP + (mb + j)]    = (f16)(dv[j] + bvn); // v transposed
        }
      }
    }

    // ---- stage 2: phi_k (one token per lane), stored transposed ----
    {
      const int t = lane;
      float s2 = 0.0f, s4 = 0.0f;
      for (int m = 0; m < D_; ++m) {
        float y = (fmaxf(ksraw[t * RP + m], 0.0f) + EPSV) * inv;
        float y2 = y * y;
        s2 += y2;
        s4 += y2 * y2;
      }
      const float r = sqrtf(s2) / sqrtf(s4);
      for (int m = 0; m < D_; ++m) {
        float y = (fmaxf(ksraw[t * RP + m], 0.0f) + EPSV) * inv;
        phiT[m * TP + t] = (f16)(r * y * y);
      }
      phiT[D_ * TP + t] = (f16)0.0f;
    }

    // ---- ksum partial (2 m-rows per lane) ----
#pragma unroll
    for (int rr = 0; rr < 2; ++rr) {
      const int m = lane + rr * 32;
      float s = 0.0f;
      for (int t = 0; t < 32; ++t) s += (float)phiT[m * TP + t];
      atomicAdd(&ksum[m], s);
    }

    // ---- stage 3: ktv += phi_k^T x v (WMMA, K = 32 tokens) ----
    v16h Am[4], Bd[4];
#pragma unroll
    for (int mt = 0; mt < 4; ++mt) Am[mt] = load_lA(phiT, TP, mt * 16, 0, lane);
#pragma unroll
    for (int dt = 0; dt < 4; ++dt) Bd[dt] = load_lB(vT, TP, dt * 16, 0, lane);
#pragma unroll
    for (int mt = 0; mt < 4; ++mt)
#pragma unroll
      for (int dt = 0; dt < 4; ++dt)
        acc[mt * 4 + dt] = wmma16(Am[mt], Bd[dt], acc[mt * 4 + dt]);
  }

  // ---- reduce into workspace ----
  float* gk = g_ktv + (size_t)(b * H_ + h) * 64 * 64;
#pragma unroll
  for (int mt = 0; mt < 4; ++mt)
#pragma unroll
    for (int dt = 0; dt < 4; ++dt) {
      v8f a = acc[mt * 4 + dt];
      const int n  = dt * 16 + (lane & 15);
      const int mb = mt * 16 + ((lane >> 4) * 8);
#pragma unroll
      for (int j = 0; j < 8; ++j) atomicAdd(&gk[(mb + j) * 64 + n], a[j]);
    }
  __syncthreads();
  if (tid < 64) atomicAdd(&g_ksum[(size_t)(b * H_ + h) * 64 + tid], ksum[tid]);
}

// =============================== Kernel C ====================================
// Per 128-token block: q projection + phi_q, numerator = phi_q x ktv,
// vss = v x vmap_w^T (all WMMA), denominator, head mean, time component.
constexpr int KC_THREADS = 128;
constexpr int KC_WAVES   = 4;

constexpr size_t C_WQT   = 0;                          // f16[64*WP]
constexpr size_t C_WVT   = C_WQT + 64 * WP * 2;
constexpr size_t C_KTVT  = C_WVT + 64 * WP * 2;
constexpr size_t C_VMAPT = C_KTVT + 64 * WP * 2;
constexpr size_t C_BQ    = C_VMAPT + 64 * WP * 2;      // float[64]
constexpr size_t C_BV    = C_BQ + 256;
constexpr size_t C_KSL   = C_BV + 256;
constexpr size_t C_VB    = C_KSL + 256;
constexpr size_t C_WAVE0 = C_VB + 256;
constexpr size_t CW_QS   = 0;                          // float[32*RP]
constexpr size_t CW_PHIQ = CW_QS + 32 * RP * 4;        // f16[32*WP]
constexpr size_t CW_VR   = CW_PHIQ + 32 * WP * 2;      // f16[32*WP]
constexpr size_t CW_OUT  = CW_VR + 32 * WP * 2;        // float[32*RP]
constexpr size_t CW_DEN  = CW_OUT + 32 * RP * 4;       // float[32]
constexpr size_t CW_SZ   = CW_DEN + 128;
constexpr size_t SMEM_C  = C_WAVE0 + KC_WAVES * CW_SZ;

__global__ __launch_bounds__(KC_THREADS)
void attn_out_kernel(const float* __restrict__ q_in,
                     const float* __restrict__ s_in,
                     const float* __restrict__ Wq, const float* __restrict__ bq_g,
                     const float* __restrict__ Wv, const float* __restrict__ bv_g,
                     const float* __restrict__ nscale,
                     const float* __restrict__ vmw, const float* __restrict__ vmb,
                     const float* __restrict__ g_ktv,
                     const float* __restrict__ g_ksum,
                     float* __restrict__ out) {
  extern __shared__ char smem[];
  f16*   WqT   = (f16*)(smem + C_WQT);
  f16*   WvT   = (f16*)(smem + C_WVT);
  f16*   ktvT  = (f16*)(smem + C_KTVT);
  f16*   vmapT = (f16*)(smem + C_VMAPT);
  float* bq    = (float*)(smem + C_BQ);
  float* bv    = (float*)(smem + C_BV);
  float* ksl   = (float*)(smem + C_KSL);
  float* vbl   = (float*)(smem + C_VB);

  const int tid  = threadIdx.x;
  const int lane = tid & 31;
  const int w    = tid >> 5;
  char* wbp      = smem + C_WAVE0 + (size_t)w * CW_SZ;
  float* qsraw   = (float*)(wbp + CW_QS);
  f16*   phiQ    = (f16*)(wbp + CW_PHIQ);
  f16*   vrow    = (f16*)(wbp + CW_VR);
  float* outb    = (float*)(wbp + CW_OUT);
  float* den_l   = (float*)(wbp + CW_DEN);

  const int bpb = N_ / 128;
  const int b   = blockIdx.x / bpb;
  const int tb  = (blockIdx.x % bpb) * 128;
  const int tok0 = tb + w * 32;
  const float inv = 1.0f / fabsf(nscale[0]);

  // vmap_w^T is already Bt-format ([d][m]); load once.
  for (int i = tid; i < 64 * 64; i += KC_THREADS) {
    int d = i >> 6, m = i & 63;
    vmapT[d * WP + m] = (f16)((d < D_ && m < D_) ? vmw[d * D_ + m] : 0.0f);
  }
  if (tid < 64) vbl[tid] = (tid < D_) ? vmb[tid] : 0.0f;
  for (int i = lane; i < 32 * RP; i += 32) outb[i] = 0.0f;

  // Hoisted activation tiles (same tokens for every head).
  v16h Aq[2][2], As[2][2];
  const float* qbase = q_in + ((size_t)b * N_ + tok0) * C_;
  const float* sbase = s_in + ((size_t)b * N_ + tok0) * C_;
#pragma unroll
  for (int rt = 0; rt < 2; ++rt) {
    const float* qr = qbase + (size_t)(rt * 16 + (lane & 15)) * C_;
    const float* sr = sbase + (size_t)(rt * 16 + (lane & 15)) * C_;
#pragma unroll
    for (int ks = 0; ks < 2; ++ks) {
      Aq[rt][ks] = load_gA(qr, ks * 32 + ((lane >> 4) * 8));
      As[rt][ks] = load_gA(sr, ks * 32 + ((lane >> 4) * 8));
    }
  }
  __syncthreads();

  for (int h = 0; h < H_; ++h) {
    __syncthreads();
    for (int i = tid; i < 64 * 64; i += KC_THREADS) {
      int o = i >> 6, c = i & 63;
      float wq = (o < D_) ? Wq[((size_t)h * D_ + o) * C_ + c] : 0.0f;
      float wv = (o < D_) ? Wv[((size_t)h * D_ + o) * C_ + c] : 0.0f;
      WqT[o * WP + c] = (f16)wq;
      WvT[o * WP + c] = (f16)wv;
      // ktvT[d][m] = ktv[m][d]
      ktvT[o * WP + c] = (f16)g_ktv[((size_t)(b * H_ + h) * 64 + c) * 64 + o];
    }
    if (tid < 64) {
      bq[tid]  = (tid < D_) ? bq_g[h * D_ + tid] : 0.0f;
      bv[tid]  = (tid < D_) ? bv_g[h * D_ + tid] : 0.0f;
      ksl[tid] = g_ksum[(size_t)(b * H_ + h) * 64 + tid];
    }
    __syncthreads();

    // ---- q and v projections ----
#pragma unroll
    for (int ct = 0; ct < 4; ++ct) {
      v16h Bq0 = load_lB(WqT, WP, ct * 16, 0, lane);
      v16h Bq1 = load_lB(WqT, WP, ct * 16, 32, lane);
      v16h Bv0 = load_lB(WvT, WP, ct * 16, 0, lane);
      v16h Bv1 = load_lB(WvT, WP, ct * 16, 32, lane);
      const int n = ct * 16 + (lane & 15);
      const float bqn = bq[n], bvn = bv[n];
#pragma unroll
      for (int rt = 0; rt < 2; ++rt) {
        v8f dq = vzero();
        dq = wmma16(Aq[rt][0], Bq0, dq);
        dq = wmma16(Aq[rt][1], Bq1, dq);
        v8f dv = vzero();
        dv = wmma16(As[rt][0], Bv0, dv);
        dv = wmma16(As[rt][1], Bv1, dv);
        const int mb = rt * 16 + ((lane >> 4) * 8);
#pragma unroll
        for (int j = 0; j < 8; ++j) {
          const int row = mb + j;
          qsraw[row * RP + n] = dq[j] + bqn;
          vrow[row * WP + n]  = (f16)(dv[j] + bvn);
        }
      }
    }

    // ---- phi_q + denominator (one token per lane) ----
    {
      const int t = lane;
      float s2 = 0.0f, s4 = 0.0f;
      for (int m = 0; m < D_; ++m) {
        float y = (fmaxf(qsraw[t * RP + m], 0.0f) + EPSV) * inv;
        float y2 = y * y;
        s2 += y2;
        s4 += y2 * y2;
      }
      const float r = sqrtf(s2) / sqrtf(s4);
      float den = 0.0f;
      for (int m = 0; m < D_; ++m) {
        float y = (fmaxf(qsraw[t * RP + m], 0.0f) + EPSV) * inv;
        float p = r * y * y;
        phiQ[t * WP + m] = (f16)p;
        den += p * ksl[m];
      }
      phiQ[t * WP + D_] = (f16)0.0f;
      den_l[t] = den;
    }

    // ---- numerator + vss via WMMA, accumulate over heads ----
#pragma unroll
    for (int rt = 0; rt < 2; ++rt) {
      v16h An0 = load_lA(phiQ, WP, rt * 16, 0, lane);
      v16h An1 = load_lA(phiQ, WP, rt * 16, 32, lane);
      v16h Av0 = load_lA(vrow, WP, rt * 16, 0, lane);
      v16h Av1 = load_lA(vrow, WP, rt * 16, 32, lane);
#pragma unroll
      for (int ct = 0; ct < 4; ++ct) {
        v16h Bn0 = load_lB(ktvT, WP, ct * 16, 0, lane);
        v16h Bn1 = load_lB(ktvT, WP, ct * 16, 32, lane);
        v16h Bm0 = load_lB(vmapT, WP, ct * 16, 0, lane);
        v16h Bm1 = load_lB(vmapT, WP, ct * 16, 32, lane);
        v8f dn = vzero();
        dn = wmma16(An0, Bn0, dn);
        dn = wmma16(An1, Bn1, dn);
        v8f dm = vzero();
        dm = wmma16(Av0, Bm0, dm);
        dm = wmma16(Av1, Bm1, dm);
        const int n = ct * 16 + (lane & 15);
        const int mb = rt * 16 + ((lane >> 4) * 8);
        const float vbn = vbl[n];
#pragma unroll
        for (int j = 0; j < 8; ++j) {
          const int row = mb + j;
          outb[row * RP + n] += dn[j] / (den_l[row] + EPSV) + dm[j] + vbn;
        }
      }
    }
  }
  __syncthreads();

  // ---- head mean + Lorentz time component, write output ----
  {
    const int t = lane;
    float* orow = out + ((size_t)b * N_ + tok0 + t) * C_;
    float ss = 0.0f;
    for (int d = 0; d < D_; ++d) {
      float ov = outb[t * RP + d] * (1.0f / (float)H_);
      ss += ov * ov;
      orow[1 + d] = ov;
    }
    orow[0] = sqrtf(ss + KCURV);
  }
}

// ================================ launcher ===================================
extern "C" void kernel_launch(void* const* d_in, const int* in_sizes, int n_in,
                              void* d_out, int out_size, void* d_ws,
                              size_t ws_size, hipStream_t stream) {
  (void)in_sizes; (void)n_in; (void)out_size; (void)ws_size;
  const float* q_in = (const float*)d_in[0];
  const float* s_in = (const float*)d_in[1];
  const float* Wq   = (const float*)d_in[2];
  const float* bq   = (const float*)d_in[3];
  const float* Wk   = (const float*)d_in[4];
  const float* bk   = (const float*)d_in[5];
  const float* Wv   = (const float*)d_in[6];
  const float* bv   = (const float*)d_in[7];
  const float* ns   = (const float*)d_in[8];
  const float* vmw  = (const float*)d_in[9];
  const float* vmb  = (const float*)d_in[10];
  float* out  = (float*)d_out;
  float* ktv  = (float*)d_ws;
  float* ksum = ktv + KTV_ELEMS;

  const int nz = KTV_ELEMS + KSUM_ELEMS;
  zero_ws_kernel<<<(nz + 255) / 256, 256, 0, stream>>>(ktv, nz);

  const int blocksA = B_ * H_ * (N_ / CHUNK);
  kv_ktv_kernel<<<blocksA, KA_THREADS, SMEM_A, stream>>>(s_in, Wk, bk, Wv, bv,
                                                         ns, ktv, ksum);

  const int blocksC = B_ * (N_ / 128);
  attn_out_kernel<<<blocksC, KC_THREADS, SMEM_C, stream>>>(
      q_in, s_in, Wq, bq, Wv, bv, ns, vmw, vmb, ktv, ksum, out);
}
